// SparseMoE_54795192763068
// MI455X (gfx1250) — compile-verified, matrix operand order checked by reading, not run on hardware
//
#include <hip/hip_runtime.h>
#include <hip/hip_bf16.h>

// ---------------------------------------------------------------------------
// Sparse MoE (top-2 of 8 experts), B=4 S=2048 D=1024 H=4096, on gfx1250.
// bf16 WMMA (v_wmma_f32_16x16x32_bf16) with f32 accumulation.
// TILE_M=32: every weight B-fragment is reused across two M sub-tiles,
// halving L2 weight traffic vs the 16-token tile (AI: 16 -> 32 flop/byte).
// ---------------------------------------------------------------------------

typedef __attribute__((ext_vector_type(16))) __bf16 v16bf;
typedef __attribute__((ext_vector_type(8)))  float  v8f;

union FragU { uint4 q[2]; v16bf v; };

constexpr int BQ = 4, SQ = 2048, DQ = 1024, EQ = 8, HQ = 4096;
constexpr int TOKENS = BQ * SQ;          // 8192
constexpr int TILE_M = 32;               // tokens per expert tile (2 WMMA M sub-tiles)
constexpr int TILES  = TOKENS / TILE_M;  // 256 (worst case: all tokens -> 1 expert)
constexpr int HCHUNK = 128;              // H processed per LDS round-trip
constexpr int XSTR   = DQ + 8;           // padded LDS strides (16B pad -> no bank conflict)
constexpr int HSTR   = HCHUNK + 8;

// workspace layout (bytes)
constexpr size_t SZ_WT   = (size_t)EQ * DQ * HQ * 2;     // 64 MB each
constexpr size_t OFF_W1T = 0;
constexpr size_t OFF_W2T = SZ_WT;
constexpr size_t OFF_CNT = 2 * SZ_WT;
constexpr size_t OFF_LST = OFF_CNT + 256;
constexpr size_t OFF_WGT = OFF_LST + (size_t)EQ * TOKENS * sizeof(int);

__device__ __forceinline__ unsigned short f2bf(float x) {
  union { float f; unsigned int u; } c; c.f = x;
  unsigned int u = c.u;
  u += 0x7FFFu + ((u >> 16) & 1u);      // round-to-nearest-even
  return (unsigned short)(u >> 16);
}

// Load one 16x32 bf16 WMMA operand fragment (A or B; row index = lane&15,
// K halves per ISA layout: lanes16-31 take K+8 / K+24). Two b128 loads/lane.
__device__ __forceinline__ v16bf load_frag(const unsigned short* base,
                                           size_t row_stride, int kbase) {
  const int lane = threadIdx.x & 31;
  const unsigned short* p = base + (size_t)(lane & 15) * row_stride + kbase
                          + ((lane & 16) ? 8 : 0);
  FragU f;
  f.q[0] = *reinterpret_cast<const uint4*>(p);       // K .. K+7
  f.q[1] = *reinterpret_cast<const uint4*>(p + 16);  // K+16 .. K+23
  return f.v;
}

// ---------------------------------------------------------------------------
// 1) f32 -> bf16 weight conversion with transpose so WMMA B-fragments read
//    contiguous K:  W1t[e][h][d] , W2t[e][d][h]
// ---------------------------------------------------------------------------
__global__ void convert_weights_kernel(const float* __restrict__ W1,
                                       const float* __restrict__ W2,
                                       unsigned short* __restrict__ W1t,
                                       unsigned short* __restrict__ W2t) {
  const size_t N = (size_t)EQ * DQ * HQ;
  const size_t stride = (size_t)gridDim.x * blockDim.x;
  for (size_t i = (size_t)blockIdx.x * blockDim.x + threadIdx.x; i < N; i += stride) {
    size_t e   = i / ((size_t)HQ * DQ);
    size_t rem = i - e * (size_t)HQ * DQ;
    size_t h  = rem / DQ, d  = rem - h * DQ;            // W1t index decomposition
    W1t[i] = f2bf(W1[(e * DQ + d) * HQ + h]);
    size_t d2 = rem / HQ, h2 = rem - d2 * HQ;           // W2t index decomposition
    W2t[i] = f2bf(W2[(e * HQ + h2) * DQ + d2]);
  }
}

// ---------------------------------------------------------------------------
// 2) Router: one wave32 per token. logits = x @ Wr + br ; top-2 ; 2-way
//    softmax (the NEG_INF terms underflow to 0, so softmax reduces to top-2).
//    Compacts (token, weight) into per-expert lists.
// ---------------------------------------------------------------------------
__global__ void router_kernel(const float* __restrict__ x,
                              const float* __restrict__ Wr,
                              const float* __restrict__ br,
                              int*   __restrict__ counts,
                              int*   __restrict__ lists,
                              float* __restrict__ wlist) {
  const int wave = threadIdx.x >> 5;
  const int lane = threadIdx.x & 31;
  const int t = blockIdx.x * (blockDim.x >> 5) + wave;
  if (t >= TOKENS) return;

  float acc[EQ];
#pragma unroll
  for (int e = 0; e < EQ; ++e) acc[e] = 0.f;

  const float* xr = x + (size_t)t * DQ;
  for (int d = lane; d < DQ; d += 32) {
    const float xv = xr[d];
    const float* wr = Wr + (size_t)d * EQ;   // 8 consecutive floats
#pragma unroll
    for (int e = 0; e < EQ; ++e) acc[e] += xv * wr[e];
  }
#pragma unroll
  for (int e = 0; e < EQ; ++e)
    for (int off = 16; off > 0; off >>= 1) acc[e] += __shfl_xor(acc[e], off, 32);

  if (lane == 0) {
    float lg[EQ];
#pragma unroll
    for (int e = 0; e < EQ; ++e) lg[e] = acc[e] + br[e];
    int i0 = 0;
#pragma unroll
    for (int e = 1; e < EQ; ++e) if (lg[e] > lg[i0]) i0 = e;   // strict > == lowest-index tie-break
    int i1 = (i0 == 0) ? 1 : 0;
#pragma unroll
    for (int e = 0; e < EQ; ++e) if (e != i0 && lg[e] > lg[i1]) i1 = e;
    const float dlt = __expf(lg[i1] - lg[i0]);
    const float z  = 1.f + dlt;
    const float w0 = 1.f / z, w1 = dlt / z;

    int p0 = atomicAdd(&counts[i0], 1);
    lists[i0 * TOKENS + p0] = t; wlist[i0 * TOKENS + p0] = w0;
    int p1 = atomicAdd(&counts[i1], 1);
    lists[i1 * TOKENS + p1] = t; wlist[i1 * TOKENS + p1] = w1;
  }
}

// ---------------------------------------------------------------------------
// 3) Fused expert MLP: out[t] += w * (relu(x[t]@W1e + b1e) @ W2e + b2e)
//    Block = 32 tokens of one expert, 8 waves. x-tile bf16 in LDS; H is
//    processed in 128-wide chunks bounced through LDS; each wave owns a
//    128-wide D slice of the f32 output accumulators (16 WMMA C-tiles:
//    2 M sub-tiles x 8 N tiles; each weight B-fragment is used twice).
// ---------------------------------------------------------------------------
__global__ __launch_bounds__(256)
void moe_expert_kernel(const float* __restrict__ x,
                       const float* __restrict__ b1,
                       const float* __restrict__ b2,
                       const unsigned short* __restrict__ W1t,
                       const unsigned short* __restrict__ W2t,
                       const int*   __restrict__ counts,
                       const int*   __restrict__ lists,
                       const float* __restrict__ wlist,
                       float* __restrict__ out) {
  __shared__ __align__(16) unsigned short xlds[TILE_M * XSTR];  // 66 KB
  __shared__ __align__(16) unsigned short hlds[TILE_M * HSTR];  // 8.7 KB
  __shared__ int   stok[TILE_M];
  __shared__ float swt [TILE_M];

  const int e    = blockIdx.y;
  const int cnt  = counts[e];
  const int row0 = blockIdx.x * TILE_M;
  if (row0 >= cnt) return;                         // inactive tail tile
  const int mTok = (cnt - row0 < TILE_M) ? (cnt - row0) : TILE_M;

  const int tid  = threadIdx.x;
  const int wave = tid >> 5;
  const int lane = tid & 31;
  const int n    = lane & 15;

  if (tid < TILE_M) {
    if (tid < mTok) {
      stok[tid] = lists[e * TOKENS + row0 + tid];
      swt [tid] = wlist[e * TOKENS + row0 + tid];
    } else { stok[tid] = -1; swt[tid] = 0.f; }
  }
  __syncthreads();

  // gather + downconvert x tile (coalesced along D)
  for (int i = tid; i < TILE_M * DQ; i += 256) {
    const int m = i >> 10, d = i & (DQ - 1);
    const int t = stok[m];
    float v = (t >= 0) ? x[(size_t)t * DQ + d] : 0.f;
    xlds[m * XSTR + d] = f2bf(v);
  }
  __syncthreads();

  v8f accC[16];                       // [p*8 + t]: p = M sub-tile, t = N tile
  const v8f vzero = {};
#pragma unroll
  for (int t = 0; t < 16; ++t) accC[t] = vzero;

  for (int hc = 0; hc < HQ / HCHUNK; ++hc) {
    // ---- phase A: h32x16 = relu(x_tile @ W1[:, wave's 16 H cols] + b1) ----
    v8f acch0 = vzero, acch1 = vzero;
    const unsigned short* w1base =
        W1t + ((size_t)e * HQ + hc * HCHUNK + wave * 16) * DQ;
    for (int kd = 0; kd < DQ; kd += 32) {
      v16bf b  = load_frag(w1base, DQ, kd);              // shared across both M tiles
      v16bf a0 = load_frag(xlds,                XSTR, kd);
      v16bf a1 = load_frag(xlds + 16 * XSTR,    XSTR, kd);
      acch0 = __builtin_amdgcn_wmma_f32_16x16x32_bf16(
          false, a0, false, b, (short)0, acch0, false, false);
      acch1 = __builtin_amdgcn_wmma_f32_16x16x32_bf16(
          false, a1, false, b, (short)0, acch1, false, false);
    }
    if (hc + 1 < HQ / HCHUNK) {   // pull next W1 chunk toward L2 (global_prefetch_b8)
      const unsigned short* nxt =
          W1t + ((size_t)e * HQ + (hc + 1) * HCHUNK + wave * 16 + n) * DQ;
      __builtin_prefetch(nxt, 0, 0);
    }

    const float bias = b1[(size_t)e * HQ + hc * HCHUNK + wave * 16 + n];
    __syncthreads();                       // prev chunk's hlds fully consumed
#pragma unroll
    for (int r = 0; r < 8; ++r) {          // C layout: M = r + 8*(lane>=16), N = lane&15
      const int m = r + ((lane >> 4) << 3);
      hlds[m * HSTR + wave * 16 + n]            = f2bf(fmaxf(acch0[r] + bias, 0.f));
      hlds[(m + 16) * HSTR + wave * 16 + n]     = f2bf(fmaxf(acch1[r] + bias, 0.f));
    }
    __syncthreads();

    // ---- phase B: out_slice += h_chunk @ W2[chunk, wave's 128 D cols] -----
    for (int kh = 0; kh < HCHUNK; kh += 32) {
      v16bf a0 = load_frag(hlds,             HSTR, kh);
      v16bf a1 = load_frag(hlds + 16 * HSTR, HSTR, kh);
#pragma unroll
      for (int t = 0; t < 8; ++t) {
        const unsigned short* w2base =
            W2t + ((size_t)e * DQ + wave * 128 + t * 16) * HQ + (size_t)hc * HCHUNK;
        v16bf b = load_frag(w2base, HQ, kh);             // used by both M tiles
        accC[t] = __builtin_amdgcn_wmma_f32_16x16x32_bf16(
            false, a0, false, b, (short)0, accC[t], false, false);
        accC[8 + t] = __builtin_amdgcn_wmma_f32_16x16x32_bf16(
            false, a1, false, b, (short)0, accC[8 + t], false, false);
      }
    }
  }

  // ---- epilogue: (y + b2) * routing_weight, atomic f32 add into out -------
#pragma unroll
  for (int t = 0; t < 8; ++t) {
    const int dcol = wave * 128 + t * 16 + n;
    const float bz = b2[(size_t)e * DQ + dcol];
#pragma unroll
    for (int p = 0; p < 2; ++p) {
#pragma unroll
      for (int r = 0; r < 8; ++r) {
        const int m = p * 16 + r + ((lane >> 4) << 3);
        if (m < mTok) {
          const float val = (accC[p * 8 + t][r] + bz) * swt[m];
          atomicAdd(&out[(size_t)stok[m] * DQ + dcol], val);
        }
      }
    }
  }
}

// ---------------------------------------------------------------------------
extern "C" void kernel_launch(void* const* d_in, const int* in_sizes, int n_in,
                              void* d_out, int out_size, void* d_ws, size_t ws_size,
                              hipStream_t stream) {
  const float* x  = (const float*)d_in[0];
  const float* Wr = (const float*)d_in[1];
  const float* br = (const float*)d_in[2];
  const float* W1 = (const float*)d_in[3];
  const float* b1 = (const float*)d_in[4];
  const float* W2 = (const float*)d_in[5];
  const float* b2 = (const float*)d_in[6];
  float* out = (float*)d_out;

  char* ws = (char*)d_ws;
  unsigned short* W1t = (unsigned short*)(ws + OFF_W1T);
  unsigned short* W2t = (unsigned short*)(ws + OFF_W2T);
  int*   counts = (int*)  (ws + OFF_CNT);
  int*   lists  = (int*)  (ws + OFF_LST);
  float* wlist  = (float*)(ws + OFF_WGT);

  hipMemsetAsync(out, 0, (size_t)out_size * sizeof(float), stream);
  hipMemsetAsync(counts, 0, 256, stream);

  convert_weights_kernel<<<4096, 256, 0, stream>>>(W1, W2, W1t, W2t);
  router_kernel<<<TOKENS / 8, 256, 0, stream>>>(x, Wr, br, counts, lists, wlist);

  dim3 grid(TILES, EQ);
  moe_expert_kernel<<<grid, 256, 0, stream>>>(x, b1, b2, W1t, W2t,
                                              counts, lists, wlist, out);
}
